// AdapterLayer_18442589569221
// MI455X (gfx1250) — compile-verified, matrix operand order checked by reading, not run on hardware
//
#include <hip/hip_runtime.h>

// ---------------------------------------------------------------------------
// MoE FFT-attention adapter for MI455X (gfx1250, wave32, WMMA bf16 + TDM).
// All 1x1 convs -> v_wmma_f32_16x16x32_bf16 GEMMs on channel-last bf16.
// 8x8 patch circular conv stages its tiles into LDS via tensor_load_to_lds.
// Gated early-exit per (batch, expert) using on-device top-2 routing.
// ---------------------------------------------------------------------------

typedef __bf16 bf16;
typedef __attribute__((ext_vector_type(16))) __bf16 v16bf;
typedef __attribute__((ext_vector_type(8)))  __bf16 v8bf;
typedef __attribute__((ext_vector_type(8)))  float  v8f;
typedef __attribute__((ext_vector_type(4)))  unsigned int v4u;
typedef __attribute__((ext_vector_type(8)))  int v8i;
typedef __attribute__((ext_vector_type(4)))  int v4i;

#define BATCH 16
#define NEXP  4
#define DIMC  64
#define RANKC 32
#define HW    16384          // 128*128
#define NTOT  (BATCH * HW)   // 262144 spatial positions

// ---------------- fragment loaders (ISA 16-bit A/B layout) -----------------
// lane<16: row/col = lane,    K chunks {k0..k0+7} and {k0+16..k0+23}
// lane>=16: row/col = lane-16, K chunks {k0+8..k0+15} and {k0+24..k0+31}

__device__ inline v16bf frag_from_chunks(v8bf lo, v8bf hi) {
    v16bf f;
#pragma unroll
    for (int i = 0; i < 8; ++i) { f[i] = lo[i]; f[i + 8] = hi[i]; }
    return f;
}

// activations / any row-major bf16 [rows, KT]: fragment rows r0..r0+15, ks k0..k0+31
__device__ inline v16bf load_frag_bf16(const bf16* __restrict__ base, int r0,
                                       int KT, int k0, int lane) {
    const int half = lane >> 4;
    const int r    = r0 + (lane & 15);
    const bf16* p  = base + (size_t)r * KT + k0 + half * 8;
    v8bf lo = *(const v8bf*)p;
    v8bf hi = *(const v8bf*)(p + 16);
    return frag_from_chunks(lo, hi);
}

// fp32 weight matrix [rows, KT], converted to bf16 fragment on the fly
__device__ inline v16bf load_frag_wf32(const float* __restrict__ W, int r0,
                                       int KT, int k0, int lane) {
    const int half = lane >> 4;
    const int r    = r0 + (lane & 15);
    const float* p = W + (size_t)r * KT + k0 + half * 8;
    float4 a = *(const float4*)p;
    float4 b = *(const float4*)(p + 4);
    float4 c = *(const float4*)(p + 16);
    float4 d = *(const float4*)(p + 20);
    v16bf f;
    f[0]  = (bf16)a.x; f[1]  = (bf16)a.y; f[2]  = (bf16)a.z; f[3]  = (bf16)a.w;
    f[4]  = (bf16)b.x; f[5]  = (bf16)b.y; f[6]  = (bf16)b.z; f[7]  = (bf16)b.w;
    f[8]  = (bf16)c.x; f[9]  = (bf16)c.y; f[10] = (bf16)c.z; f[11] = (bf16)c.w;
    f[12] = (bf16)d.x; f[13] = (bf16)d.y; f[14] = (bf16)d.z; f[15] = (bf16)d.w;
    return f;
}

// ---------------- Tensor Data Mover: 2D tile -> LDS ------------------------
// D# group0: count=1, lds_addr, 57-bit global_addr, type=2 ("image").
// D# group1: data_size=1 (2 bytes), huge tensor dims (no OOB clipping),
//            tile_dim0 elements per row, tile_dim1 rows, dim0 row stride.
#if defined(__has_builtin)
#if __has_builtin(__builtin_amdgcn_tensor_load_to_lds) && \
    __has_builtin(__builtin_amdgcn_s_wait_tensorcnt)
#define HAVE_TDM 1
#endif
#endif
#ifndef HAVE_TDM
#define HAVE_TDM 0
#endif

#if HAVE_TDM
__device__ inline void tdm_load_2d_bf16(const void* gptr, unsigned lds_byte_off,
                                        unsigned tile_d0_elems, unsigned tile_d1,
                                        unsigned dim0_stride_elems) {
    const unsigned long long ga = (unsigned long long)(size_t)gptr;
    v4u g0;
    g0[0] = 1u;                                   // count=1, user descriptor
    g0[1] = lds_byte_off;                         // LDS destination (bytes)
    g0[2] = (unsigned)ga;                         // global_addr[31:0]
    g0[3] = (unsigned)((ga >> 32) & 0x01FFFFFFu)  // global_addr[56:32]
            | 0x80000000u;                        // type = 2 ("image")
    v8i g1;
    g1[0] = 0x00010000;                           // data_size=1 (2B), no mask/pad
    g1[1] = 0;                                    // atomic_barrier_addr=0, dim0 lo16=0
    g1[2] = 0x10;                                 // tensor_dim0=0x100000, dim1 lo16=0
    g1[3] = (int)((tile_d0_elems << 16) | 0x10u); // tile_dim0 | tensor_dim1 hi16
    g1[4] = (int)tile_d1;                         // tile_dim1 (tile_dim2=0)
    g1[5] = (int)dim0_stride_elems;               // tensor_dim0_stride lo32
    g1[6] = 0;                                    // stride hi16, dim1_stride lo16
    g1[7] = 0;
    v4i z4 = {0, 0, 0, 0};
#if __clang_major__ >= 23
    v8i z8 = {0, 0, 0, 0, 0, 0, 0, 0};
    __builtin_amdgcn_tensor_load_to_lds(g0, g1, z4, z4, z8, 0);
#else
    __builtin_amdgcn_tensor_load_to_lds(g0, g1, z4, z4, 0);
#endif
}
#endif

// ---------------- routing -------------------------------------------------
__global__ __launch_bounds__(256) void pool_mean_kernel(
    const float* __restrict__ x, float* __restrict__ pooled) {
    __shared__ float red[256];
    const int bc   = blockIdx.x;                // b*64 + c, 0..1023
    const float* p = x + ((size_t)bc << 14);
    float s = 0.f;
    for (int i = threadIdx.x; i < HW; i += 256) s += p[i];
    red[threadIdx.x] = s;
    __syncthreads();
    for (int st = 128; st > 0; st >>= 1) {
        if (threadIdx.x < st) red[threadIdx.x] += red[threadIdx.x + st];
        __syncthreads();
    }
    if (threadIdx.x == 0) pooled[bc] = red[0] * (1.f / (float)HW);
}

__global__ void routing_kernel(const float* __restrict__ pooled,
                               const float* __restrict__ freq_emb,
                               const float* __restrict__ noise,
                               const float* __restrict__ gate_w,
                               const float* __restrict__ fgate_w,
                               float* __restrict__ gates,
                               float* __restrict__ gsum) {
    const int b = threadIdx.x;
    if (b >= BATCH) return;
    float lg[NEXP];
#pragma unroll
    for (int e = 0; e < NEXP; ++e) {
        float s = 0.f;
        for (int c = 0; c < DIMC; ++c) s += pooled[b * DIMC + c] * gate_w[e * DIMC + c];
        for (int f = 0; f < 64; ++f)   s += freq_emb[b * 64 + f] * fgate_w[e * 64 + f];
        lg[e] = s + noise[b * NEXP + e] * (1.f / (float)NEXP);
    }
    float mx = lg[0];
#pragma unroll
    for (int e = 1; e < NEXP; ++e) mx = fmaxf(mx, lg[e]);
    float p[NEXP], tot = 0.f;
#pragma unroll
    for (int e = 0; e < NEXP; ++e) { p[e] = __expf(lg[e] - mx); tot += p[e]; }
#pragma unroll
    for (int e = 0; e < NEXP; ++e) p[e] /= tot;
    int i1 = 0;
#pragma unroll
    for (int e = 1; e < NEXP; ++e) if (p[e] > p[i1]) i1 = e;
    int i2 = -1;
#pragma unroll
    for (int e = 0; e < NEXP; ++e)
        if (e != i1 && (i2 < 0 || p[e] > p[i2])) i2 = e;
#pragma unroll
    for (int e = 0; e < NEXP; ++e)
        gates[b * NEXP + e] = (e == i1 || e == i2) ? p[e] : 0.f;
    gsum[b] = p[i1] + p[i2];
}

// ---------------- NCHW fp32 -> channel-last bf16 transpose -----------------
__global__ __launch_bounds__(256) void transpose_cl_kernel(
    const float* __restrict__ in, bf16* __restrict__ out) {
    __shared__ float tile[64][65];
    const int t  = threadIdx.x;
    const int b  = blockIdx.x >> 8;          // 256 blocks per batch image
    const int m0 = (blockIdx.x & 255) * 64;  // 64 positions per block
    {
        const int c  = t >> 2;
        const int ms = (t & 3) * 16;
        const float* src = in + (((size_t)b * 64 + c) << 14) + m0 + ms;
#pragma unroll
        for (int i = 0; i < 4; ++i) {
            float4 v = *(const float4*)(src + i * 4);
            tile[c][ms + i * 4 + 0] = v.x;
            tile[c][ms + i * 4 + 1] = v.y;
            tile[c][ms + i * 4 + 2] = v.z;
            tile[c][ms + i * 4 + 3] = v.w;
        }
    }
    __syncthreads();
    {
        const int m  = t >> 2;
        const int cs = (t & 3) * 16;
        bf16* dst = out + ((size_t)(b * HW + m0 + m)) * 64 + cs;
        v8bf o0, o1;
#pragma unroll
        for (int j = 0; j < 8; ++j) o0[j] = (bf16)tile[cs + j][m];
#pragma unroll
        for (int j = 0; j < 8; ++j) o1[j] = (bf16)tile[cs + 8 + j][m];
        *(v8bf*)dst       = o0;
        *(v8bf*)(dst + 8) = o1;
    }
}

// ---------------- out = gatesum[b] * x  (fp32 NCHW) ------------------------
__global__ __launch_bounds__(256) void init_out_kernel(
    const float* __restrict__ x, const float* __restrict__ gsum,
    float* __restrict__ out) {
    const size_t i = (size_t)blockIdx.x * 256 + threadIdx.x;  // float4 index
    const int b    = (int)(i >> 18);                          // 2^18 float4 per image
    const float g  = gsum[b];
    float4 v = ((const float4*)x)[i];
    v.x *= g; v.y *= g; v.z *= g; v.w *= g;
    ((float4*)out)[i] = v;
}

// ---------------- generic channel-last WMMA GEMM ---------------------------
// out[m, n] = sum_k act[m, k] * W[n, k]      (A = weights, B = activations)
// Optional: +bias[n], silu, elementwise multiply by mulbuf[m, n].
template <int KT, int NOUT, bool SILU, bool HASBIAS, bool HASMUL>
__global__ __launch_bounds__(256) void gemm_cl_kernel(
    const bf16* __restrict__ act, const float* __restrict__ W,
    const float* __restrict__ bias, const bf16* __restrict__ mulbuf,
    bf16* __restrict__ out, const float* __restrict__ gates, int e) {
    const int lane = threadIdx.x & 31;
    const int wave = threadIdx.x >> 5;
    const int tile = blockIdx.x * 8 + wave;
    const int m0   = tile * 16;
    const int b    = m0 >> 14;
    if (gates[b * NEXP + e] == 0.f) return;   // expert inactive for this image

    // pull the activation tile toward the caches while weights convert
    __builtin_prefetch(act + (size_t)m0 * KT, 0, 3);

    constexpr int KC = KT / 32;
    constexpr int NT = NOUT / 16;

    v16bf wf[NT][KC];
#pragma unroll
    for (int nt = 0; nt < NT; ++nt)
#pragma unroll
        for (int kc = 0; kc < KC; ++kc)
            wf[nt][kc] = load_frag_wf32(W, nt * 16, KT, kc * 32, lane);

    v8f acc[NT] = {};
#pragma unroll
    for (int kc = 0; kc < KC; ++kc) {
        v16bf bfrag = load_frag_bf16(act, m0, KT, kc * 32, lane);
#pragma unroll
        for (int nt = 0; nt < NT; ++nt)
            acc[nt] = __builtin_amdgcn_wmma_f32_16x16x32_bf16(
                false, wf[nt][kc], false, bfrag, (short)0, acc[nt], false, false);
    }

    const int sp    = m0 + (lane & 15);       // spatial index for this lane
    const int chalf = (lane >> 4) * 8;        // cout sub-offset
#pragma unroll
    for (int nt = 0; nt < NT; ++nt) {
        const int cb = nt * 16 + chalf;
        float v[8];
#pragma unroll
        for (int r = 0; r < 8; ++r) v[r] = acc[nt][r];
        if (HASBIAS) {
#pragma unroll
            for (int r = 0; r < 8; ++r) v[r] += bias[cb + r];
        }
        if (SILU) {
#pragma unroll
            for (int r = 0; r < 8; ++r) v[r] = v[r] / (1.f + __expf(-v[r]));
        }
        if (HASMUL) {
            v8bf mv = *(const v8bf*)(mulbuf + (size_t)sp * NOUT + cb);
#pragma unroll
            for (int r = 0; r < 8; ++r) v[r] *= (float)mv[r];
        }
        v8bf o;
#pragma unroll
        for (int r = 0; r < 8; ++r) o[r] = (bf16)v[r];
        *(v8bf*)(out + (size_t)sp * NOUT + cb) = o;
    }
}

// ---------------- final GEMM: NCHW fp32 gated accumulation -----------------
// out[b, cout, hw] += gate * sum_k act[m, k] * p2[cout, k]
__global__ __launch_bounds__(256) void gemm_p2_acc_kernel(
    const bf16* __restrict__ act, const float* __restrict__ W,
    float* __restrict__ out, const float* __restrict__ gates, int e) {
    const int lane = threadIdx.x & 31;
    const int wave = threadIdx.x >> 5;
    const int tile = blockIdx.x * 8 + wave;
    const int m0   = tile * 16;
    const int b    = m0 >> 14;
    const int hw0  = m0 & (HW - 1);
    const float g  = gates[b * NEXP + e];
    if (g == 0.f) return;

    v16bf af = load_frag_bf16(act, m0, 32, 0, lane);   // A = activations (M=spatial)
#pragma unroll
    for (int nt = 0; nt < 4; ++nt) {                   // 4 cout tiles of 16
        v16bf wfr = load_frag_wf32(W, nt * 16, 32, 0, lane);  // B = weights (N=cout)
        v8f acc = {};
        acc = __builtin_amdgcn_wmma_f32_16x16x32_bf16(
            false, af, false, wfr, (short)0, acc, false, false);
        const int cout = nt * 16 + (lane & 15);
        const int hwb  = hw0 + (lane >> 4) * 8;
        float* p = out + (((size_t)b * DIMC + cout) << 14) + hwb;
        float4 o0 = *(float4*)p;
        float4 o1 = *(float4*)(p + 4);
        o0.x += g * acc[0]; o0.y += g * acc[1]; o0.z += g * acc[2]; o0.w += g * acc[3];
        o1.x += g * acc[4]; o1.y += g * acc[5]; o1.z += g * acc[6]; o1.w += g * acc[7];
        *(float4*)p       = o0;
        *(float4*)(p + 4) = o1;
    }
}

// ---------------- depthwise conv (channel-last bf16, zero padding) ---------
template <int C, int R>
__global__ __launch_bounds__(256) void dwconv_cl_kernel(
    const bf16* __restrict__ in, const float* __restrict__ w,
    bf16* __restrict__ out, const float* __restrict__ gates, int e) {
    constexpr int CSH = (C == 32) ? 5 : 6;
    constexpr int D   = 2 * R + 1;
    const int idx = blockIdx.x * 256 + threadIdx.x;
    const int c   = idx & (C - 1);
    const int m   = idx >> CSH;
    const int b   = m >> 14;
    if (gates[b * NEXP + e] == 0.f) return;
    const int h = (m >> 7) & 127;
    const int x = m & 127;
    const float* wc = w + c * D * D;
    float acc = 0.f;
#pragma unroll
    for (int dy = -R; dy <= R; ++dy) {
        const int hh = h + dy;
        if ((unsigned)hh >= 128u) continue;
#pragma unroll
        for (int dx = -R; dx <= R; ++dx) {
            const int ww = x + dx;
            if ((unsigned)ww >= 128u) continue;
            const int mm = (b << 14) | (hh << 7) | ww;
            acc += (float)in[(size_t)mm * C + c] * wc[(dy + R) * D + (dx + R)];
        }
    }
    out[(size_t)m * C + c] = (bf16)acc;
}

// ---------------- 8x8 per-patch circular convolution -----------------------
// out[u,v,c] = sum_{i,j} q[i,j,c] * k[(u-i)&7, (v-j)&7, c]
// (equivalent to irfft2(rfft2(q) * rfft2(k)) on 8x8 patches)
// q / kv patch tiles are DMA'd into LDS by the Tensor Data Mover.
__global__ __launch_bounds__(256) void circconv8_kernel(
    const bf16* __restrict__ q, const bf16* __restrict__ kv,
    bf16* __restrict__ attc, const float* __restrict__ gates, int e) {
    const int b  = blockIdx.x >> 8;
    const int pp = blockIdx.x & 255;
    if (gates[b * NEXP + e] == 0.f) return;
    const int ph = pp >> 4, pw = pp & 15;
    __shared__ bf16  q_raw[64 * 32];   // 8 rows x (8 pos x 32 ch)
    __shared__ bf16  kv_raw[64 * 64];  // 8 rows x (8 pos x 64 ch)
    __shared__ float qs[64][32];
    __shared__ float ks[64][32];
    const int t     = threadIdx.x;
    const int mbase = (b << 14) + (ph * 8) * 128 + pw * 8;

#if HAVE_TDM
    if (t < 32) {   // one wave drives the TDM; TENSORcnt tracks completion
        tdm_load_2d_bf16(q + (size_t)mbase * 32,
                         (unsigned)(size_t)&q_raw[0],
                         /*tile_d0=*/8 * 32, /*tile_d1=*/8,
                         /*row stride=*/128 * 32);
        tdm_load_2d_bf16(kv + (size_t)mbase * 64,
                         (unsigned)(size_t)&kv_raw[0],
                         /*tile_d0=*/8 * 64, /*tile_d1=*/8,
                         /*row stride=*/128 * 64);
        __builtin_amdgcn_s_wait_tensorcnt(0);
    }
    __syncthreads();
#else
#pragma unroll
    for (int r = 0; r < 8; ++r) {
        const int idx = r * 256 + t;
        const int pix = idx >> 5, c = idx & 31;
        const int u = pix >> 3, v = pix & 7;
        const int m = mbase + u * 128 + v;
        q_raw[pix * 32 + c]  = q[(size_t)m * 32 + c];
        kv_raw[pix * 64 + c] = kv[(size_t)m * 64 + c];
    }
    __syncthreads();
#endif

    // stage bf16 tiles to f32 LDS for the MAC loops
#pragma unroll
    for (int r = 0; r < 8; ++r) {
        const int idx = r * 256 + t;            // 0..2047
        const int pix = idx >> 5, c = idx & 31;
        qs[pix][c] = (float)q_raw[idx];
        ks[pix][c] = (float)kv_raw[pix * 64 + c];   // k = first 32 channels
    }
    __syncthreads();

    const int c = t & 31;
    const int u = t >> 5;          // this thread's output row (0..7)
    float acc[8] = {};
    for (int i = 0; i < 8; ++i) {
        const int iu = (u - i) & 7;
#pragma unroll
        for (int j = 0; j < 8; ++j) {
            const float qv = qs[i * 8 + j][c];
#pragma unroll
            for (int v = 0; v < 8; ++v)
                acc[v] += qv * ks[iu * 8 + ((v - j) & 7)][c];
        }
    }
#pragma unroll
    for (int v = 0; v < 8; ++v)
        attc[(size_t)(mbase + u * 128 + v) * 32 + c] = (bf16)acc[v];
}

// ---------------- channel LayerNorm * v  (one wave == 32 channels) ---------
__global__ __launch_bounds__(256) void lnv_kernel(
    const bf16* __restrict__ attc, const bf16* __restrict__ kv,
    const float* __restrict__ lnw, const float* __restrict__ lnb,
    bf16* __restrict__ attv, const float* __restrict__ gates, int e) {
    const int gid = blockIdx.x * 256 + threadIdx.x;
    const int m   = gid >> 5;
    const int c   = gid & 31;
    const int b   = m >> 14;
    if (gates[b * NEXP + e] == 0.f) return;
    float val = (float)attc[(size_t)m * 32 + c];
    float s = val;
#pragma unroll
    for (int off = 16; off > 0; off >>= 1) s += __shfl_xor(s, off, 32);
    const float mu = s * (1.f / 32.f);
    const float d  = val - mu;
    float s2 = d * d;
#pragma unroll
    for (int off = 16; off > 0; off >>= 1) s2 += __shfl_xor(s2, off, 32);
    const float var = s2 * (1.f / 32.f);
    float y = d * rsqrtf(var + 1e-5f) * lnw[c] + lnb[c];
    y *= (float)kv[(size_t)m * 64 + 32 + c];          // v = last 32 channels
    attv[(size_t)m * 32 + c] = (bf16)y;
}

// ---------------------------------------------------------------------------
extern "C" void kernel_launch(void* const* d_in, const int* in_sizes, int n_in,
                              void* d_out, int out_size, void* d_ws, size_t ws_size,
                              hipStream_t stream) {
    (void)in_sizes; (void)n_in; (void)out_size; (void)ws_size;
    const float* x       = (const float*)d_in[0];
    const float* shared  = (const float*)d_in[1];
    const float* freq    = (const float*)d_in[2];
    const float* noise   = (const float*)d_in[3];
    const float* gate_w  = (const float*)d_in[4];
    const float* fgate_w = (const float*)d_in[5];
    const float* p0      = (const float*)d_in[6];
    const float* p1      = (const float*)d_in[7];
    const float* p2      = (const float*)d_in[8];
    const float* q_w     = (const float*)d_in[9];
    const float* q_dw    = (const float*)d_in[10];
    const float* kv_w    = (const float*)d_in[11];
    const float* kv_dw   = (const float*)d_in[12];
    const float* ln_w    = (const float*)d_in[13];
    const float* ln_b    = (const float*)d_in[14];
    const float* po_w    = (const float*)d_in[15];
    const float* po_b    = (const float*)d_in[16];
    float* out = (float*)d_out;

    // workspace carving (256-byte aligned)
    char*  ws = (char*)d_ws;
    size_t o  = 0;
    auto carve = [&](size_t bytes) -> char* {
        char* p = ws + o;
        o += (bytes + 255) & ~(size_t)255;
        return p;
    };
    bf16*  x_t    = (bf16*)carve((size_t)NTOT * 64 * 2);
    bf16*  sh_t   = (bf16*)carve((size_t)NTOT * 64 * 2);
    bf16*  xr     = (bf16*)carve((size_t)NTOT * 32 * 2);
    bf16*  qp     = (bf16*)carve((size_t)NTOT * 32 * 2);  // reused as attv
    bf16*  qb     = (bf16*)carve((size_t)NTOT * 32 * 2);  // reused as g
    bf16*  kvp    = (bf16*)carve((size_t)NTOT * 64 * 2);
    bf16*  kvb    = (bf16*)carve((size_t)NTOT * 64 * 2);
    bf16*  attc   = (bf16*)carve((size_t)NTOT * 32 * 2);
    float* pooled = (float*)carve(BATCH * DIMC * 4);
    float* gates  = (float*)carve(BATCH * NEXP * 4);
    float* gsum   = (float*)carve(BATCH * 4);
    bf16* attv = qp;   // qp dead after dw3
    bf16* gbuf = qb;   // qb dead after circconv
    bf16* hbuf = xr;   // xr dead after kv GEMM

    const int GEMM_GRID = NTOT / 16 / 8;   // 8 waves/block, 16 rows/wave

    // routing + layout transforms + residual/gate-scaled init of out
    pool_mean_kernel<<<BATCH * DIMC, 256, 0, stream>>>(x, pooled);
    routing_kernel<<<1, 32, 0, stream>>>(pooled, freq, noise, gate_w, fgate_w,
                                         gates, gsum);
    transpose_cl_kernel<<<BATCH * 256, 256, 0, stream>>>(x, x_t);
    transpose_cl_kernel<<<BATCH * 256, 256, 0, stream>>>(shared, sh_t);
    init_out_kernel<<<(NTOT * 64 / 4) / 256, 256, 0, stream>>>(x, gsum, out);

    for (int e = 0; e < NEXP; ++e) {
        const float* p0_e  = p0 + e * RANKC * DIMC;
        const float* p1_e  = p1 + e * RANKC * DIMC;
        const float* p2_e  = p2 + e * DIMC * RANKC;
        const float* qw_e  = q_w + e * RANKC * RANKC;
        const float* qdw_e = q_dw + e * RANKC * 9;
        const float* kvw_e = kv_w + e * 2 * RANKC * RANKC;
        const float* kdw_e = kv_dw + e * 2 * RANKC * 49;
        const float* lnw_e = ln_w + e * RANKC;
        const float* lnb_e = ln_b + e * RANKC;
        const float* pow_e = po_w + e * RANKC * RANKC;
        const float* pob_e = po_b + e * RANKC;

        // xr = conv1x1(x, p0)                       [64 -> 32]
        gemm_cl_kernel<64, 32, false, false, false>
            <<<GEMM_GRID, 256, 0, stream>>>(x_t, p0_e, nullptr, nullptr, xr, gates, e);
        // qp = conv1x1(xr, q_w)                     [32 -> 32]
        gemm_cl_kernel<32, 32, false, false, false>
            <<<GEMM_GRID, 256, 0, stream>>>(xr, qw_e, nullptr, nullptr, qp, gates, e);
        // qb = dwconv3x3(qp)
        dwconv_cl_kernel<32, 1><<<NTOT * 32 / 256, 256, 0, stream>>>(qp, qdw_e, qb, gates, e);
        // kvp = conv1x1(xr, kv_w)                   [32 -> 64]
        gemm_cl_kernel<32, 64, false, false, false>
            <<<GEMM_GRID, 256, 0, stream>>>(xr, kvw_e, nullptr, nullptr, kvp, gates, e);
        // kvb = dwconv7x7(kvp)
        dwconv_cl_kernel<64, 3><<<NTOT * 64 / 256, 256, 0, stream>>>(kvp, kdw_e, kvb, gates, e);
        // attc = per-patch 8x8 circular conv (== irfft2(rfft2(q)*rfft2(k)))
        circconv8_kernel<<<BATCH * 256, 256, 0, stream>>>(qb, kvb, attc, gates, e);
        // attv = LayerNorm_c(attc) * v
        lnv_kernel<<<NTOT * 32 / 256, 256, 0, stream>>>(attc, kvb, lnw_e, lnb_e,
                                                        attv, gates, e);
        // g = silu(conv1x1(shared, p1))             [64 -> 32]
        gemm_cl_kernel<64, 32, true, false, false>
            <<<GEMM_GRID, 256, 0, stream>>>(sh_t, p1_e, nullptr, nullptr, gbuf, gates, e);
        // h = (conv1x1(attv, po_w) + po_b) * g      [32 -> 32]
        gemm_cl_kernel<32, 32, false, true, true>
            <<<GEMM_GRID, 256, 0, stream>>>(attv, pow_e, pob_e, gbuf, hbuf, gates, e);
        // out += gate[b,e] * conv1x1(h, p2)         [32 -> 64], NCHW fp32
        gemm_p2_acc_kernel<<<GEMM_GRID, 256, 0, stream>>>(hbuf, p2_e, out, gates, e);
    }
}
